// MambaAutoencoder_11854109736993
// MI455X (gfx1250) — matrix operand order repaired and checked by reading, not compiled, not run on hardware
//
#include <hip/hip_runtime.h>
#include <hip/hip_bf16.h>

// ---------------------------------------------------------------------------
// Mamba-style selective SSM for MI455X (gfx1250), wave32 + WMMA f16->f32.
//   - GEMMs: 16x64 tile per wave, 4 accumulators, A-fragment reuse, K-pipelined
//   - Scan: 1 wave per (batch,channel), 2 states/thread, shuffle-only reduce
// ---------------------------------------------------------------------------

typedef __attribute__((ext_vector_type(16))) _Float16 v16h;
typedef __attribute__((ext_vector_type(8)))  _Float16 v8h;
typedef __attribute__((ext_vector_type(8)))  float    v8f;

#define B_SZ    2
#define SEQ_L   2048
#define D_MODEL 128
#define D_INNER 256
#define D_STATE 64
#define ROWS    (B_SZ * SEQ_L)      // 4096
#define N_CAT   192                 // 129 + 32 + 1 padded to x64

// ---------------- elementwise helpers --------------------------------------

__global__ void k_f32_to_f16(const float* __restrict__ src,
                             _Float16* __restrict__ dst, int n) {
  int i = blockIdx.x * blockDim.x + threadIdx.x;
  if (i < n) dst[i] = (_Float16)src[i];
}

__global__ void k_neg_exp(const float* __restrict__ src,
                          float* __restrict__ dst, int n) {
  int i = blockIdx.x * blockDim.x + threadIdx.x;
  if (i < n) dst[i] = -__expf(src[i]);
}

// pack [x_proj_w(129x256); theta_proj_w(32x256); lambda_proj_w(1x256); pad] -> f16
__global__ void k_pack_wcat(const float* __restrict__ xpw,
                            const float* __restrict__ thw,
                            const float* __restrict__ lpw,
                            _Float16* __restrict__ wcat) {
  int i = blockIdx.x * blockDim.x + threadIdx.x;   // over 192*256
  if (i >= N_CAT * D_INNER) return;
  int row = i / D_INNER, d = i % D_INNER;
  float v = 0.f;
  if (row < 129)       v = xpw[row * D_INNER + d];
  else if (row < 161)  v = thw[(row - 129) * D_INNER + d];
  else if (row == 161) v = lpw[d];
  wcat[i] = (_Float16)v;
}

// ---------------- WMMA GEMM: C[M,N] = A[M,K] * W[N,K]^T --------------------
// One wave computes a 16x64 output tile (4 x 16x16 accumulators); the A
// fragment is loaded once per K-step and reused across the 4 WMMAs.
// K multiple of 32, M multiple of 16, N multiple of 64.

__global__ void __launch_bounds__(32)
k_wmma_gemm(const _Float16* __restrict__ A, const _Float16* __restrict__ W,
            float* __restrict__ C, int M, int N, int K) {
  const int tile_m = blockIdx.x * 16;
  const int tile_n = blockIdx.y * 64;
  const int lane = threadIdx.x;        // 0..31
  const int l15  = lane & 15;
  const int hi   = lane >> 4;          // 0 or 1

  // A fragment: row = tile_m + l15; lane<16 -> K {0..7,16..23}, lane>=16 -> K {8..15,24..31}
  const _Float16* arow = A + (long)(tile_m + l15) * K + hi * 8;
  // B fragments: col = tile_n + 16*j + l15; lane<16 -> K {0..7,16..23}, lane>=16 -> K {8..15,24..31}
  const _Float16* bcol0 = W + (long)(tile_n +  0 + l15) * K + hi * 8;
  const _Float16* bcol1 = W + (long)(tile_n + 16 + l15) * K + hi * 8;
  const _Float16* bcol2 = W + (long)(tile_n + 32 + l15) * K + hi * 8;
  const _Float16* bcol3 = W + (long)(tile_n + 48 + l15) * K + hi * 8;

  v8f acc0 = {}, acc1 = {}, acc2 = {}, acc3 = {};
  for (int kb = 0; kb < K; kb += 32) {
    __builtin_prefetch(arow + kb + 32, 0, 1);
    v8h a0 = *(const v8h*)(arow + kb);
    v8h a1 = *(const v8h*)(arow + kb + 16);
    v16h a;
#pragma unroll
    for (int i = 0; i < 8; ++i) { a[i] = a0[i]; a[i + 8] = a1[i]; }

#define LOAD_B(bc)                                                     \
    ({ v8h b0_ = *(const v8h*)((bc) + kb);                             \
       v8h b1_ = *(const v8h*)((bc) + kb + 16);                        \
       v16h b_;                                                        \
       _Pragma("unroll")                                               \
       for (int i = 0; i < 8; ++i) { b_[i] = b0_[i]; b_[i+8] = b1_[i]; } \
       b_; })

    v16h b0 = LOAD_B(bcol0);
    v16h b1 = LOAD_B(bcol1);
    v16h b2 = LOAD_B(bcol2);
    v16h b3 = LOAD_B(bcol3);
#undef LOAD_B
    acc0 = __builtin_amdgcn_wmma_f32_16x16x32_f16(false, a, false, b0, (short)0, acc0, false, false);
    acc1 = __builtin_amdgcn_wmma_f32_16x16x32_f16(false, a, false, b1, (short)0, acc1, false, false);
    acc2 = __builtin_amdgcn_wmma_f32_16x16x32_f16(false, a, false, b2, (short)0, acc2, false, false);
    acc3 = __builtin_amdgcn_wmma_f32_16x16x32_f16(false, a, false, b3, (short)0, acc3, false, false);
  }
#pragma unroll
  for (int r = 0; r < 8; ++r) {
    long row = tile_m + r + hi * 8;
    C[row * N + tile_n +  0 + l15] = acc0[r];
    C[row * N + tile_n + 16 + l15] = acc1[r];
    C[row * N + tile_n + 32 + l15] = acc2[r];
    C[row * N + tile_n + 48 + l15] = acc3[r];
  }
}

// ---------------- depthwise causal conv (D_CONV=4) + SiLU ------------------
// x_ssm branch is xz cols [0,256); row stride of xz is 512. Emits f32 + f16.

__global__ void k_conv_silu(const float* __restrict__ xz,
                            const float* __restrict__ cw,
                            const float* __restrict__ cb,
                            float* __restrict__ xs,
                            _Float16* __restrict__ xs16) {
  int i = blockIdx.x * blockDim.x + threadIdx.x;   // over 4096*256
  if (i >= ROWS * D_INNER) return;
  int r = i / D_INNER, c = i % D_INNER;
  int b = r / SEQ_L, l = r % SEQ_L;
  float acc = cb[c];
#pragma unroll
  for (int j = 0; j < 4; ++j) {
    int ll = l - 3 + j;
    if (ll >= 0) acc += cw[c * 4 + j] * xz[(long)(b * SEQ_L + ll) * 512 + c];
  }
  float v = acc / (1.f + __expf(-acc));   // SiLU
  xs[i] = v;
  xs16[i] = (_Float16)v;
}

// ---------------- theta cumsum over sequence -------------------------------
// params layout: [row][192]; theta at cols 129..160. One thread per (b, j).

__global__ void k_theta_cumsum(const float* __restrict__ params,
                               float* __restrict__ tc) {
  int t = threadIdx.x;                 // 0..63
  int b = t >> 5, j = t & 31;
  float run = 0.f;
  for (int l = 0; l < SEQ_L; ++l) {
    long r = (long)b * SEQ_L + l;
    run += params[r * N_CAT + 129 + j];
    tc[r * 32 + j] = run;
  }
}

// ---------------- RoPE on (B + bias) and (C + bias) ------------------------

__global__ void k_rope_bc(const float* __restrict__ params,
                          const float* __restrict__ tc,
                          const float* __restrict__ bb,
                          const float* __restrict__ cbias,
                          float* __restrict__ Bc, float* __restrict__ Cc) {
  int i = blockIdx.x * blockDim.x + threadIdx.x;   // over 4096*64
  if (i >= ROWS * D_STATE) return;
  int r = i / D_STATE, j = i % D_STATE;
  float th = tc[(long)r * 32 + (j & 31)];
  float cs = cosf(th), sn = sinf(th);
  const float* p = params + (long)r * N_CAT;
  float bv = p[j] + bb[j];
  float br = (j < 32) ? -(p[j + 32] + bb[j + 32]) : (p[j - 32] + bb[j - 32]);
  Bc[i] = bv * cs + br * sn;
  float cv = p[64 + j] + cbias[j];
  float cr = (j < 32) ? -(p[64 + j + 32] + cbias[j + 32])
                      : (p[64 + j - 32] + cbias[j - 32]);
  Cc[i] = cv * cs + cr * sn;
}

// ---------------- delta = softplus(dt*w+b), lam = sigmoid(...) -------------

__global__ void k_delta_lam(const float* __restrict__ params,
                            const float* __restrict__ dtw,
                            const float* __restrict__ dtb,
                            const float* __restrict__ lamb,
                            float* __restrict__ delta,
                            float* __restrict__ lam) {
  int i = blockIdx.x * blockDim.x + threadIdx.x;   // over 4096*256
  if (i >= ROWS * D_INNER) return;
  int r = i / D_INNER, c = i % D_INNER;
  float dt = params[(long)r * N_CAT + 128];
  float v = dt * dtw[c] + dtb[c];
  delta[i] = (v > 20.f) ? v : log1pf(__expf(v));
  if (c == 0) {
    float lv = params[(long)r * N_CAT + 161] + lamb[0];
    lam[r] = 1.f / (1.f + __expf(-lv));
  }
}

// ---------------- SSM sequential scan --------------------------------------
// One wave (32 threads) per (batch, channel); each thread owns the adjacent
// state pair (2t, 2t+1) so B/C loads are b64 and the sum over 64 states is a
// pure shfl_xor tree: no LDS, no barriers in the 2048-step hot loop.
// Fuses the D skip connection and the SiLU(z) gate; writes y in f16.

__global__ void __launch_bounds__(32)
k_ssm_scan(const float* __restrict__ Bc, const float* __restrict__ Cc,
           const float* __restrict__ xs, const float* __restrict__ delta,
           const float* __restrict__ lam, const float* __restrict__ Aneg,
           const float* __restrict__ xz,  const float* __restrict__ Dp,
           _Float16* __restrict__ y16) {
  int b = blockIdx.x / D_INNER;
  int c = blockIdx.x % D_INNER;
  int t = threadIdx.x;                 // owns states 2t, 2t+1
  float2 A2 = ((const float2*)(Aneg + c * D_STATE))[t];
  float Dc = Dp[c];
  float h0 = 0.f, h1 = 0.f, bxp0 = 0.f, bxp1 = 0.f;

  for (int l = 0; l < SEQ_L; ++l) {
    long r = (long)b * SEQ_L + l;
    float2 bv = ((const float2*)(Bc + r * D_STATE))[t];
    float2 cv = ((const float2*)(Cc + r * D_STATE))[t];
    float xv = xs[r * D_INNER + c];
    float dl = delta[r * D_INNER + c];
    float lm = lam[r];
    float a0 = __expf(dl * A2.x);
    float a1 = __expf(dl * A2.y);
    float bx0 = bv.x * xv;
    float bx1 = bv.y * xv;
    h0 = a0 * h0 + lm * dl * bx0 + (1.f - lm) * dl * a0 * bxp0;
    h1 = a1 * h1 + lm * dl * bx1 + (1.f - lm) * dl * a1 * bxp1;
    bxp0 = bx0; bxp1 = bx1;

    float p = h0 * cv.x + h1 * cv.y;
#pragma unroll
    for (int off = 16; off >= 1; off >>= 1) p += __shfl_xor(p, off, 32);
    if (t == 0) {
      float yv = p + xv * Dc;
      float zv = xz[r * 512 + 256 + c];
      float g = zv / (1.f + __expf(-zv));   // SiLU(z)
      y16[r * D_INNER + c] = (_Float16)(yv * g);
    }
  }
}

// ---------------------------------------------------------------------------

extern "C" void kernel_launch(void* const* d_in, const int* in_sizes, int n_in,
                              void* d_out, int out_size, void* d_ws, size_t ws_size,
                              hipStream_t stream) {
  const float* x        = (const float*)d_in[0];
  const float* in_w     = (const float*)d_in[1];
  const float* conv_w   = (const float*)d_in[2];
  const float* conv_b   = (const float*)d_in[3];
  const float* xproj_w  = (const float*)d_in[4];
  const float* dt_w     = (const float*)d_in[5];
  const float* dt_b     = (const float*)d_in[6];
  const float* A_log    = (const float*)d_in[7];
  const float* B_bias   = (const float*)d_in[8];
  const float* C_bias   = (const float*)d_in[9];
  const float* th_w     = (const float*)d_in[10];
  const float* lp_w     = (const float*)d_in[11];
  const float* lp_b     = (const float*)d_in[12];
  const float* out_w    = (const float*)d_in[13];
  const float* D_param  = (const float*)d_in[14];
  float* out = (float*)d_out;

  // workspace carve-out (all sizes multiples of 256B)
  char* p = (char*)d_ws;
  _Float16* x_f16   = (_Float16*)p;  p += (size_t)ROWS * D_MODEL * 2;      // 1.0 MB
  _Float16* w_in16  = (_Float16*)p;  p += (size_t)512 * 128 * 2;           // 128 KB
  _Float16* w_cat16 = (_Float16*)p;  p += (size_t)N_CAT * D_INNER * 2;     // 96 KB
  _Float16* w_out16 = (_Float16*)p;  p += (size_t)128 * 256 * 2;           // 64 KB
  float*    A_neg   = (float*)p;     p += (size_t)D_INNER * D_STATE * 4;   // 64 KB
  float*    xz      = (float*)p;     p += (size_t)ROWS * 512 * 4;          // 8 MB
  float*    xs      = (float*)p;     p += (size_t)ROWS * D_INNER * 4;      // 4 MB
  _Float16* xs_f16  = (_Float16*)p;  p += (size_t)ROWS * D_INNER * 2;      // 2 MB
  float*    params  = (float*)p;     p += (size_t)ROWS * N_CAT * 4;        // 3 MB
  float*    th_cum  = (float*)p;     p += (size_t)ROWS * 32 * 4;           // 0.5 MB
  float*    Bc      = (float*)p;     p += (size_t)ROWS * D_STATE * 4;      // 1 MB
  float*    Cc      = (float*)p;     p += (size_t)ROWS * D_STATE * 4;      // 1 MB
  float*    delta   = (float*)p;     p += (size_t)ROWS * D_INNER * 4;      // 4 MB
  float*    lamv    = (float*)p;     p += (size_t)ROWS * 4;                // 16 KB
  _Float16* y_f16   = (_Float16*)p;  p += (size_t)ROWS * D_INNER * 2;      // 2 MB
  (void)ws_size; (void)n_in; (void)in_sizes; (void)out_size;

  const int TB = 256;
  auto blk = [](int n, int t) { return (n + t - 1) / t; };

  // 1. conversions / precompute
  k_f32_to_f16<<<blk(ROWS * D_MODEL, TB), TB, 0, stream>>>(x, x_f16, ROWS * D_MODEL);
  k_f32_to_f16<<<blk(512 * 128, TB), TB, 0, stream>>>(in_w, w_in16, 512 * 128);
  k_pack_wcat<<<blk(N_CAT * D_INNER, TB), TB, 0, stream>>>(xproj_w, th_w, lp_w, w_cat16);
  k_f32_to_f16<<<blk(128 * 256, TB), TB, 0, stream>>>(out_w, w_out16, 128 * 256);
  k_neg_exp<<<blk(D_INNER * D_STATE, TB), TB, 0, stream>>>(A_log, A_neg, D_INNER * D_STATE);

  // 2. in_proj GEMM: 4096x128 @ (512x128)^T -> 4096x512
  k_wmma_gemm<<<dim3(ROWS / 16, 512 / 64), 32, 0, stream>>>(
      x_f16, w_in16, xz, ROWS, 512, 128);

  // 3. depthwise conv + SiLU (emits f32 + f16)
  k_conv_silu<<<blk(ROWS * D_INNER, TB), TB, 0, stream>>>(
      xz, conv_w, conv_b, xs, xs_f16);

  // 4. fused x_proj/theta/lambda GEMM: 4096x256 @ (192x256)^T -> 4096x192
  k_wmma_gemm<<<dim3(ROWS / 16, N_CAT / 64), 32, 0, stream>>>(
      xs_f16, w_cat16, params, ROWS, N_CAT, D_INNER);

  // 5. cumsum, RoPE, delta/lam
  k_theta_cumsum<<<1, 64, 0, stream>>>(params, th_cum);
  k_rope_bc<<<blk(ROWS * D_STATE, TB), TB, 0, stream>>>(
      params, th_cum, B_bias, C_bias, Bc, Cc);
  k_delta_lam<<<blk(ROWS * D_INNER, TB), TB, 0, stream>>>(
      params, dt_w, dt_b, lp_b, delta, lamv);

  // 6. SSM scan (fused D skip + SiLU(z) gate), writes y in f16
  k_ssm_scan<<<B_SZ * D_INNER, 32, 0, stream>>>(
      Bc, Cc, xs, delta, lamv, A_neg, xz, D_param, y_f16);

  // 7. out_proj GEMM: 4096x256 @ (128x256)^T -> 4096x128 (final output)
  k_wmma_gemm<<<dim3(ROWS / 16, D_MODEL / 64), 32, 0, stream>>>(
      y_f16, w_out16, out, ROWS, D_MODEL, D_INNER);
}
